// MTMDModel_54030688583964
// MI455X (gfx1250) — matrix operand order repaired and checked by reading, not compiled, not run on hardware
//
#include <hip/hip_runtime.h>

// ---------------------------------------------------------------------------
// MTMD forward for MI455X (gfx1250): bf16 WMMA GEMMs + async global->LDS
// staging + fp32 epilogues.
// ---------------------------------------------------------------------------

typedef __bf16 bf16;
typedef __attribute__((ext_vector_type(16))) __bf16 v16bf;
typedef __attribute__((ext_vector_type(8)))  __bf16 v8bf;
typedef __attribute__((ext_vector_type(8)))  float  v8f;

#define NBATCH 4096
#define TSTEPS 60
#define DIN    6
#define HID    128
#define CCON   128
#define MMEM   128
#define GATE3  384   // 3*HID
#define TOPK3  3

#define GEMM_AB  0   // C = A(MxK) * B(KxN), both row-major
#define GEMM_ABT 1   // C = A(MxK) * Bt(NxK)^T
#define GEMM_ATB 2   // C = At(KxM)^T * B(KxN)

static inline int cdiv(int a, int b) { return (a + b - 1) / b; }

// 16-byte-per-lane async copy global -> LDS (ASYNCcnt tracked, no VGPR data).
#define ASYNC_LDS_B128(ldsaddr, gaddr)                                        \
    asm volatile("global_load_async_to_lds_b128 %0, %1, off"                  \
                 :: "v"(ldsaddr), "v"(gaddr) : "memory")
#define ASYNC_WAIT() asm volatile("s_wait_asynccnt 0" ::: "memory")

// ---------------------------------------------------------------------------
// WMMA GEMM: 128x64 tile per block, 256 threads = 8 waves.
// Wave w owns row-tile w (16 rows) x 4 column accumulators -> 4 WMMAs per
// A-fragment per 32-wide K step. M % 128 == 0, N % 64 == 0, K % 32 == 0.
// ---------------------------------------------------------------------------
template <int MODE>
__global__ __launch_bounds__(256)
void gemm_wmma(const bf16* __restrict__ A, const bf16* __restrict__ B,
               float* __restrict__ C, bf16* __restrict__ Cbf,
               const float* __restrict__ bias,
               int Mdim, int Ndim, int Kdim, int act, int accum)
{
    __shared__ __align__(64) bf16 As[128][32];  // As[m][k]
    __shared__ __align__(64) bf16 Bs[64][32];   // Bs[n][k]

    const int mblk = blockIdx.y * 128;
    const int nblk = blockIdx.x * 64;
    const int tid  = threadIdx.x;
    const int lane = tid & 31;
    const int wave = tid >> 5;                  // row tile 0..7

    v8f acc[4] = {{}, {}, {}, {}};

    for (int k0 = 0; k0 < Kdim; k0 += 32) {
        // ---- stage A tile (128 x 32) ----
        if constexpr (MODE == GEMM_ATB) {
            // A given as At (K x M): strided gather
            int k  = tid >> 3;                  // 0..31
            int mo = (tid & 7) * 16;            // 0..112
            const bf16* src = &A[(size_t)(k0 + k) * Mdim + (mblk + mo)];
            __builtin_prefetch(src + 32 * Mdim, 0, 1);
            #pragma unroll
            for (int j = 0; j < 16; ++j)
                As[mo + j][k] = src[j];
        } else {
            // row-major A: two contiguous 16B async copies per thread
            int m  = tid >> 1;                  // 0..127
            int ko = (tid & 1) * 16;            // 0 or 16
            const bf16* src = &A[(size_t)(mblk + m) * Kdim + (k0 + ko)];
            unsigned l0 = (unsigned)(uintptr_t)&As[m][ko];
            unsigned l1 = (unsigned)(uintptr_t)&As[m][ko + 8];
            unsigned long long g0 = (unsigned long long)src;
            unsigned long long g1 = (unsigned long long)(src + 8);
            ASYNC_LDS_B128(l0, g0);
            ASYNC_LDS_B128(l1, g1);
        }
        // ---- stage B tile as Bs[n][k] (64 x 32) ----
        if constexpr (MODE == GEMM_ABT) {
            // B given as Bt (N x K): contiguous 16B async copy per thread
            int n  = tid >> 2;                  // 0..63
            int ko = (tid & 3) * 8;             // 0..24
            const bf16* src = &B[(size_t)(nblk + n) * Kdim + (k0 + ko)];
            unsigned l0 = (unsigned)(uintptr_t)&Bs[n][ko];
            unsigned long long g0 = (unsigned long long)src;
            ASYNC_LDS_B128(l0, g0);
        } else {
            // B row-major (K x N): strided gather
            int k  = tid >> 3;                  // 0..31
            int no = (tid & 7) * 8;             // 0..56
            const bf16* src = &B[(size_t)(k0 + k) * Ndim + (nblk + no)];
            __builtin_prefetch(src + 32 * Ndim, 0, 1);
            #pragma unroll
            for (int j = 0; j < 8; ++j)
                Bs[no + j][k] = src[j];
        }
        ASYNC_WAIT();
        __syncthreads();

        // ---- A fragment in contiguous k order (B uses the same) ----
        // lanes 0-15: M=lane,    K in {0..15}
        // lanes16-31: M=lane-16, K in {16..31}
        int mrow = wave * 16 + (lane & 15);
        int kA   = (lane >= 16) ? 16 : 0;
        v8bf alo = *(const v8bf*)&As[mrow][kA];
        v8bf ahi = *(const v8bf*)&As[mrow][kA + 8];
        v16bf afrag = __builtin_shufflevector(alo, ahi,
                        0,1,2,3,4,5,6,7,8,9,10,11,12,13,14,15);

        // ---- preload all 4 B fragments, then issue WMMAs back-to-back ----
        int kB = (lane >= 16) ? 16 : 0;
        v16bf bfr[4];
        #pragma unroll
        for (int ct = 0; ct < 4; ++ct)
            bfr[ct] = *(const v16bf*)&Bs[ct * 16 + (lane & 15)][kB];
        #pragma unroll
        for (int ct = 0; ct < 4; ++ct)
            acc[ct] = __builtin_amdgcn_wmma_f32_16x16x32_bf16(
                          false, afrag, false, bfr[ct], (short)0, acc[ct],
                          false, false);
        __syncthreads();
    }

    // ---- epilogue per ISA C/D layout: VGPR r -> M = r (+8 for lanes>=16) ----
    const int mbase = mblk + wave * 16 + ((lane >= 16) ? 8 : 0);
    if (accum) {
        #pragma unroll
        for (int ct = 0; ct < 4; ++ct) {
            int ncol = nblk + ct * 16 + (lane & 15);
            float bs = bias ? bias[ncol] : 0.f;
            #pragma unroll
            for (int r = 0; r < 8; ++r) {
                float v = acc[ct][r] + bs;
                if (act == 1) v = (v > 0.f) ? v : 0.01f * v;
                size_t off = (size_t)(mbase + r) * Ndim + ncol;
                v += C[off];
                C[off] = v;
                if (Cbf) Cbf[off] = (bf16)v;
            }
        }
    } else {
        #pragma unroll
        for (int ct = 0; ct < 4; ++ct) {
            int ncol = nblk + ct * 16 + (lane & 15);
            float bs = bias ? bias[ncol] : 0.f;
            #pragma unroll
            for (int r = 0; r < 8; ++r) {
                float v = acc[ct][r] + bs;
                if (act == 1) v = (v > 0.f) ? v : 0.01f * v;
                size_t off = (size_t)(mbase + r) * Ndim + ncol;
                C[off] = v;
                if (Cbf) Cbf[off] = (bf16)v;
            }
        }
    }
}

// ---------------------------------------------------------------------------
// Small kernels
// ---------------------------------------------------------------------------
__global__ void zero_f32(float* p, int n) {
    int i = blockIdx.x * blockDim.x + threadIdx.x;
    if (i < n) p[i] = 0.f;
}
__global__ void zero_bf(bf16* p, int n) {
    int i = blockIdx.x * blockDim.x + threadIdx.x;
    if (i < n) p[i] = (bf16)0.f;
}
__global__ void f2bf(const float* s, bf16* d, int n) {
    int i = blockIdx.x * blockDim.x + threadIdx.x;
    if (i < n) d[i] = (bf16)s[i];
}
__global__ void ew_sub(const float* a, const float* b, float* o, bf16* obf, int n) {
    int i = blockIdx.x * blockDim.x + threadIdx.x;
    if (i < n) { float v = a[i] - b[i]; o[i] = v; if (obf) obf[i] = (bf16)v; }
}
__global__ void ew_sub3(const float* a, const float* b, const float* c,
                        float* o, bf16* obf, int n) {
    int i = blockIdx.x * blockDim.x + threadIdx.x;
    if (i < n) { float v = a[i] - b[i] - c[i]; o[i] = v; if (obf) obf[i] = (bf16)v; }
}
__global__ void ew_mul_inplace(float* p, const float* r, bf16* pbf, int n) {
    int i = blockIdx.x * blockDim.x + threadIdx.x;
    if (i < n) { float v = p[i] * r[i]; p[i] = v; pbf[i] = (bf16)v; }
}

// layer-0 GRU input projection: K = DIN = 6 (too small for WMMA; VALU)
__global__ void gi0_kernel(const float* __restrict__ x, const float* __restrict__ W,
                           const float* __restrict__ b, float* __restrict__ GI, int t) {
    int i = blockIdx.x * blockDim.x + threadIdx.x;
    if (i >= NBATCH * GATE3) return;
    int n = i / GATE3, c = i % GATE3;
    float s = b[c];
    #pragma unroll
    for (int d = 0; d < DIN; ++d)
        s += x[(size_t)n * (DIN * TSTEPS) + d * TSTEPS + t] * W[c * DIN + d];
    GI[i] = s;
}

__global__ void gru_gate(const float* __restrict__ GI, const float* __restrict__ GH,
                         float* __restrict__ h, bf16* __restrict__ hbf,
                         bf16* __restrict__ seq_out) {
    int i = blockIdx.x * blockDim.x + threadIdx.x;
    if (i >= NBATCH * HID) return;
    int n = i / HID, hh = i % HID;
    size_t base = (size_t)n * GATE3;
    float ir = GI[base + hh], iz = GI[base + HID + hh], in_ = GI[base + 2 * HID + hh];
    float hr = GH[base + hh], hz = GH[base + HID + hh], hn  = GH[base + 2 * HID + hh];
    float r  = 1.f / (1.f + expf(-(ir + hr)));
    float z  = 1.f / (1.f + expf(-(iz + hz)));
    float nc = tanhf(in_ + r * hn);
    float hv = (1.f - z) * nc + z * h[i];
    h[i] = hv;
    hbf[i] = (bf16)hv;
    if (seq_out) seq_out[i] = (bf16)hv;
}

__global__ __launch_bounds__(256)
void col_sum(const float* S, float* out, int R, int Ccols) {
    int c = blockIdx.x;
    __shared__ float red[256];
    float s = 0.f;
    for (int r = threadIdx.x; r < R; r += 256) s += S[(size_t)r * Ccols + c];
    red[threadIdx.x] = s; __syncthreads();
    for (int st = 128; st > 0; st >>= 1) {
        if (threadIdx.x < st) red[threadIdx.x] += red[threadIdx.x + st];
        __syncthreads();
    }
    if (threadIdx.x == 0) out[c] = red[0];
}

__global__ __launch_bounds__(256)
void col_max(const float* S, float* out, int R, int Ccols) {
    int c = blockIdx.x;
    __shared__ float red[256];
    float m = -INFINITY;
    for (int r = threadIdx.x; r < R; r += 256) m = fmaxf(m, S[(size_t)r * Ccols + c]);
    red[threadIdx.x] = m; __syncthreads();
    for (int st = 128; st > 0; st >>= 1) {
        if (threadIdx.x < st) red[threadIdx.x] = fmaxf(red[threadIdx.x], red[threadIdx.x + st]);
        __syncthreads();
    }
    if (threadIdx.x == 0) out[c] = red[0];
}

__global__ __launch_bounds__(256)
void col_softmax(const float* S, float* Q, int R, int Ccols) {
    int c = blockIdx.x;
    __shared__ float red[256];
    float m = -INFINITY;
    for (int r = threadIdx.x; r < R; r += 256) m = fmaxf(m, S[(size_t)r * Ccols + c]);
    red[threadIdx.x] = m; __syncthreads();
    for (int st = 128; st > 0; st >>= 1) {
        if (threadIdx.x < st) red[threadIdx.x] = fmaxf(red[threadIdx.x], red[threadIdx.x + st]);
        __syncthreads();
    }
    m = red[0]; __syncthreads();
    float s = 0.f;
    for (int r = threadIdx.x; r < R; r += 256) {
        float e = expf(S[(size_t)r * Ccols + c] - m);
        Q[(size_t)r * Ccols + c] = e;
        s += e;
    }
    red[threadIdx.x] = s; __syncthreads();
    for (int st = 128; st > 0; st >>= 1) {
        if (threadIdx.x < st) red[threadIdx.x] += red[threadIdx.x + st];
        __syncthreads();
    }
    float inv = 1.f / red[0];
    for (int r = threadIdx.x; r < R; r += 256)
        Q[(size_t)r * Ccols + c] *= inv;
}

__global__ __launch_bounds__(256)
void row_softmax(float* S, const int* mask, bf16* Sbf, int R, int Ccols) {
    int r = blockIdx.x;
    __shared__ float red[256];
    float m = -INFINITY;
    for (int c = threadIdx.x; c < Ccols; c += 256) {
        float v = S[(size_t)r * Ccols + c];
        if (mask && !mask[c]) v = -INFINITY;
        m = fmaxf(m, v);
    }
    red[threadIdx.x] = m; __syncthreads();
    for (int st = 128; st > 0; st >>= 1) {
        if (threadIdx.x < st) red[threadIdx.x] = fmaxf(red[threadIdx.x], red[threadIdx.x + st]);
        __syncthreads();
    }
    m = red[0]; __syncthreads();
    float s = 0.f;
    for (int c = threadIdx.x; c < Ccols; c += 256) {
        size_t o = (size_t)r * Ccols + c;
        float v = S[o];
        if (mask && !mask[c]) v = -INFINITY;
        float e = expf(v - m);
        S[o] = e;
        s += e;
    }
    red[threadIdx.x] = s; __syncthreads();
    for (int st = 128; st > 0; st >>= 1) {
        if (threadIdx.x < st) red[threadIdx.x] += red[threadIdx.x + st];
        __syncthreads();
    }
    float inv = 1.f / red[0];
    for (int c = threadIdx.x; c < Ccols; c += 256) {
        size_t o = (size_t)r * Ccols + c;
        float v = S[o] * inv;
        S[o] = v;
        if (Sbf) Sbf[o] = (bf16)v;
    }
}

__global__ void row_argmax(const float* S, int* g, int R, int Ccols) {
    int r = blockIdx.x * blockDim.x + threadIdx.x;
    if (r >= R) return;
    float best = S[(size_t)r * Ccols];
    int bi = 0;
    for (int c = 1; c < Ccols; ++c) {
        float v = S[(size_t)r * Ccols + c];
        if (v > best) { best = v; bi = c; }
    }
    g[r] = bi;
}

__global__ void row_l2norm(const float* S, float* out, int R, int Ccols) {
    int r = blockIdx.x * blockDim.x + threadIdx.x;
    if (r >= R) return;
    float s = 0.f;
    for (int c = 0; c < Ccols; ++c) { float v = S[(size_t)r * Ccols + c]; s += v * v; }
    out[r] = sqrtf(s);
}

__global__ void row_nonzero(const float* S, int* flag, int R, int Ccols) {
    int r = blockIdx.x * blockDim.x + threadIdx.x;
    if (r >= R) return;
    float s = 0.f;
    for (int c = 0; c < Ccols; ++c) s += S[(size_t)r * Ccols + c];
    flag[r] = (s != 0.f) ? 1 : 0;
}

__global__ void cos_finish(float* XY, const float* na, const float* nb, int R, int Ccols) {
    int i = blockIdx.x * blockDim.x + threadIdx.x;
    if (i >= R * Ccols) return;
    int r = i / Ccols, c = i % Ccols;
    float d = na[r] * nb[c];
    float v = XY[i] / d;
    XY[i] = isnan(v) ? 0.f : v;
}

__global__ void s2c_kernel(const float* cm, const float* csum, bf16* out) {
    int i = blockIdx.x * blockDim.x + threadIdx.x;
    if (i >= NBATCH * CCON) return;
    int c = i % CCON;
    float v = cm[i] / (csum[c] * cm[i] + 1.f);
    out[i] = (bf16)v;
}

__global__ void topk3_kernel(const float* __restrict__ S, float* diag,
                             float* tv, int* tc, float* colsum) {
    int i = blockIdx.x * blockDim.x + threadIdx.x;
    if (i >= NBATCH) return;
    diag[i] = S[(size_t)i * NBATCH + i];
    float v0 = -INFINITY, v1 = -INFINITY, v2 = -INFINITY;
    int   c0 = 0, c1 = 0, c2 = 0;
    for (int j = 0; j < NBATCH; ++j) {
        float v = (j == i) ? 0.f : S[(size_t)i * NBATCH + j];
        if (v > v0)      { v2 = v1; c2 = c1; v1 = v0; c1 = c0; v0 = v; c0 = j; }
        else if (v > v1) { v2 = v1; c2 = c1; v1 = v;  c1 = j; }
        else if (v > v2) { v2 = v;  c2 = j; }
    }
    tv[i * 3 + 0] = v0; tv[i * 3 + 1] = v1; tv[i * 3 + 2] = v2;
    tc[i * 3 + 0] = c0; tc[i * 3 + 1] = c1; tc[i * 3 + 2] = c2;
    atomicAdd(&colsum[c0], v0);
    atomicAdd(&colsum[c1], v1);
    atomicAdd(&colsum[c2], v2);
}

// sparse  hid2 = h_m.T @ h_si  (3 nonzeros per row)
__global__ void scatter_hid2(const float* tv, const int* tc,
                             const float* h_si, float* hid2) {
    int idx = blockIdx.x * blockDim.x + threadIdx.x;
    if (idx >= NBATCH * TOPK3 * HID) return;
    int hh = idx % HID;
    int k  = (idx / HID) % TOPK3;
    int i  = idx / (HID * TOPK3);
    atomicAdd(&hid2[(size_t)tc[i * 3 + k] * HID + hh],
              tv[i * 3 + k] * h_si[(size_t)i * HID + hh]);
}
__global__ void hid2_diag(const float* colsum, const float* diag,
                          const float* h_si, float* hid2) {
    int idx = blockIdx.x * blockDim.x + threadIdx.x;
    if (idx >= NBATCH * HID) return;
    int j = idx / HID;
    if (colsum[j] != 0.f) hid2[idx] += diag[j] * h_si[idx];
}

// upload_mem pieces
__global__ void upload_w_kernel(const float* sq, const int* g, const float* mx, float* wv) {
    int i = blockIdx.x * blockDim.x + threadIdx.x;
    if (i >= NBATCH) return;
    int gi = g[i];
    wv[i] = sq[(size_t)i * MMEM + gi] / mx[gi];
}
__global__ void scatter_qu(const float* Q, const float* wv, const int* g, float* qu) {
    int idx = blockIdx.x * blockDim.x + threadIdx.x;
    if (idx >= NBATCH * HID) return;
    int i = idx / HID, hh = idx % HID;
    atomicAdd(&qu[(size_t)g[i] * HID + hh], wv[i] * Q[idx]);
}
__global__ void mem_finalize(const float* qu, const float* Km, float* out) {
    int m = blockIdx.x * blockDim.x + threadIdx.x;
    if (m >= MMEM) return;
    float s = 0.f;
    for (int hh = 0; hh < HID; ++hh) {
        float u = qu[m * HID + hh] + Km[m * HID + hh];
        s += u * u;
    }
    float nrm = fmaxf(sqrtf(s), 1e-12f);
    for (int hh = 0; hh < HID; ++hh)
        out[m * HID + hh] = (qu[m * HID + hh] + Km[m * HID + hh]) / nrm;
}

__global__ void pred_kernel(const float* all, const float* w, const float* b, float* out) {
    int i = blockIdx.x * blockDim.x + threadIdx.x;
    if (i >= NBATCH) return;
    float s = b[0];
    for (int hh = 0; hh < HID; ++hh) s += all[(size_t)i * HID + hh] * w[hh];
    out[i] = s;
}

// ---------------------------------------------------------------------------
// Host-side orchestration
// ---------------------------------------------------------------------------
static void gemm(hipStream_t st, const bf16* A, const bf16* B, float* C, bf16* Cbf,
                 const float* bias, int M_, int N_, int K_, int mode, int act, int accum) {
    dim3 g(N_ / 64, M_ / 128), b(256);
    switch (mode) {
    case GEMM_AB:
        gemm_wmma<GEMM_AB ><<<g, b, 0, st>>>(A, B, C, Cbf, bias, M_, N_, K_, act, accum); break;
    case GEMM_ABT:
        gemm_wmma<GEMM_ABT><<<g, b, 0, st>>>(A, B, C, Cbf, bias, M_, N_, K_, act, accum); break;
    default:
        gemm_wmma<GEMM_ATB><<<g, b, 0, st>>>(A, B, C, Cbf, bias, M_, N_, K_, act, accum); break;
    }
}

extern "C" void kernel_launch(void* const* d_in, const int* in_sizes, int n_in,
                              void* d_out, int out_size, void* d_ws, size_t ws_size,
                              hipStream_t stream) {
    (void)in_sizes; (void)n_in; (void)out_size; (void)ws_size;
    hipStream_t st = stream;

    // ---- inputs ----
    const float* x_in    = (const float*)d_in[0];
    const float* concept = (const float*)d_in[1];
    const float* m0      = (const float*)d_in[2];
    const float* m1      = (const float*)d_in[3];
    const float* W0ih = (const float*)d_in[4];  const float* W0hh = (const float*)d_in[5];
    const float* b0ih = (const float*)d_in[6];  const float* b0hh = (const float*)d_in[7];
    const float* W1ih = (const float*)d_in[8];  const float* W1hh = (const float*)d_in[9];
    const float* b1ih = (const float*)d_in[10]; const float* b1hh = (const float*)d_in[11];
    const float* fc_ps_w      = (const float*)d_in[12]; const float* fc_ps_b      = (const float*)d_in[13];
    const float* fc_hs_w      = (const float*)d_in[14]; const float* fc_hs_b      = (const float*)d_in[15];
    const float* fc_ps_fore_w = (const float*)d_in[16]; const float* fc_ps_fore_b = (const float*)d_in[17];
    const float* fc_hs_fore_w = (const float*)d_in[18]; const float* fc_hs_fore_b = (const float*)d_in[19];
    const float* fc_ps_back_w = (const float*)d_in[20]; const float* fc_ps_back_b = (const float*)d_in[21];
    const float* fc_hs_back_w = (const float*)d_in[22]; const float* fc_hs_back_b = (const float*)d_in[23];
    const float* fc_indi_w    = (const float*)d_in[24]; const float* fc_indi_b    = (const float*)d_in[25];
    const float* fc_out_w     = (const float*)d_in[26]; const float* fc_out_b     = (const float*)d_in[27];

    float* out_pred = (float*)d_out;
    float* out_mem0 = out_pred + NBATCH;
    float* out_mem1 = out_mem0 + MMEM * HID;

    // ---- workspace bump allocator ----
    char* wp = (char*)d_ws; size_t off = 0;
    auto alloc = [&](size_t bytes) -> void* {
        void* p = wp + off;
        off = (off + bytes + 255) & ~(size_t)255;
        return p;
    };
    bf16*  hseq0_bf = (bf16*)alloc((size_t)NBATCH * TSTEPS * HID * 2);
    float* h        = (float*)alloc((size_t)NBATCH * HID * 4);
    bf16*  h_bf     = (bf16*)alloc((size_t)NBATCH * HID * 2);
    float* GI       = (float*)alloc((size_t)NBATCH * GATE3 * 4);
    float* GH       = (float*)alloc((size_t)NBATCH * GATE3 * 4);
    bf16*  W0hh_bf  = (bf16*)alloc((size_t)GATE3 * HID * 2);
    bf16*  W1ih_bf  = (bf16*)alloc((size_t)GATE3 * HID * 2);
    bf16*  W1hh_bf  = (bf16*)alloc((size_t)GATE3 * HID * 2);
    bf16*  fcw_bf[7];
    const float* fcw_src[7] = { fc_ps_w, fc_hs_w, fc_ps_fore_w, fc_hs_fore_w,
                                fc_ps_back_w, fc_hs_back_w, fc_indi_w };
    for (int i = 0; i < 7; ++i) fcw_bf[i] = (bf16*)alloc((size_t)HID * HID * 2);
    bf16*  m0_bf    = (bf16*)alloc((size_t)MMEM * HID * 2);
    bf16*  m1_bf    = (bf16*)alloc((size_t)MMEM * HID * 2);
    float* csum     = (float*)alloc(CCON * 4);
    bf16*  s2c_bf   = (bf16*)alloc((size_t)NBATCH * CCON * 2);
    float* NC_a     = (float*)alloc((size_t)NBATCH * CCON * 4);
    float* NC_b     = (float*)alloc((size_t)NBATCH * CCON * 4);
    bf16*  NC_bf    = (bf16*)alloc((size_t)NBATCH * CCON * 2);
    float* hidden0  = (float*)alloc((size_t)CCON * HID * 4);
    bf16*  hidden0_bf = (bf16*)alloc((size_t)CCON * HID * 2);
    float* hidden1  = (float*)alloc((size_t)CCON * HID * 4);
    bf16*  hidden1_bf = (bf16*)alloc((size_t)CCON * HID * 2);
    int*   valid0   = (int*)alloc(CCON * 4);
    int*   validh   = (int*)alloc(NBATCH * 4);
    float* norm1    = (float*)alloc(NBATCH * 4);
    float* norm2    = (float*)alloc(NBATCH * 4);
    float* p_shared = (float*)alloc((size_t)NBATCH * HID * 4);
    bf16*  p_shared_bf = (bf16*)alloc((size_t)NBATCH * HID * 2);
    float* Rtmp     = (float*)alloc((size_t)NBATCH * HID * 4);
    float* p_back   = (float*)alloc((size_t)NBATCH * HID * 4);
    float* all_info = (float*)alloc((size_t)NBATCH * HID * 4);
    float* score    = (float*)alloc((size_t)NBATCH * MMEM * 4);
    bf16*  score_bf = (bf16*)alloc((size_t)NBATCH * MMEM * 2);
    float* sq       = (float*)alloc((size_t)NBATCH * MMEM * 4);
    int*   gidx     = (int*)alloc(NBATCH * 4);
    float* wv       = (float*)alloc(NBATCH * 4);
    float* mx       = (float*)alloc(MMEM * 4);
    float* qu       = (float*)alloc((size_t)MMEM * HID * 4);
    float* h_si     = (float*)alloc((size_t)NBATCH * HID * 4);
    bf16*  h_si_bf  = (bf16*)alloc((size_t)NBATCH * HID * 2);
    float* NN       = (float*)alloc((size_t)NBATCH * NBATCH * 4);
    bf16*  NN_bf    = (bf16*)alloc((size_t)NBATCH * NBATCH * 2);
    float* tv       = (float*)alloc((size_t)NBATCH * TOPK3 * 4);
    int*   tc       = (int*)alloc((size_t)NBATCH * TOPK3 * 4);
    float* colsum   = (float*)alloc(NBATCH * 4);
    float* diag     = (float*)alloc(NBATCH * 4);
    float* hid2     = (float*)alloc((size_t)NBATCH * HID * 4);
    bf16*  hid2_bf  = (bf16*)alloc((size_t)NBATCH * HID * 2);
    float* h_shared = (float*)alloc((size_t)NBATCH * HID * 4);
    bf16*  h_shared_bf = (bf16*)alloc((size_t)NBATCH * HID * 2);
    float* h_back   = (float*)alloc((size_t)NBATCH * HID * 4);
    float* tmpH     = (float*)alloc((size_t)NBATCH * HID * 4);
    bf16*  tmpH_bf  = (bf16*)alloc((size_t)NBATCH * HID * 2);

    const int NH  = NBATCH * HID;
    const int NG  = NBATCH * GATE3;
    const int B256 = 256;

    // ---- weight conversions ----
    f2bf<<<cdiv(GATE3 * HID, B256), B256, 0, st>>>(W0hh, W0hh_bf, GATE3 * HID);
    f2bf<<<cdiv(GATE3 * HID, B256), B256, 0, st>>>(W1ih, W1ih_bf, GATE3 * HID);
    f2bf<<<cdiv(GATE3 * HID, B256), B256, 0, st>>>(W1hh, W1hh_bf, GATE3 * HID);
    for (int i = 0; i < 7; ++i)
        f2bf<<<cdiv(HID * HID, B256), B256, 0, st>>>(fcw_src[i], fcw_bf[i], HID * HID);
    f2bf<<<cdiv(MMEM * HID, B256), B256, 0, st>>>(m0, m0_bf, MMEM * HID);
    f2bf<<<cdiv(MMEM * HID, B256), B256, 0, st>>>(m1, m1_bf, MMEM * HID);

    // ================= GRU layer 0 =================
    zero_f32<<<cdiv(NH, B256), B256, 0, st>>>(h, NH);
    zero_bf <<<cdiv(NH, B256), B256, 0, st>>>(h_bf, NH);
    for (int t = 0; t < TSTEPS; ++t) {
        gi0_kernel<<<cdiv(NG, B256), B256, 0, st>>>(x_in, W0ih, b0ih, GI, t);
        gemm(st, h_bf, W0hh_bf, GH, nullptr, b0hh, NBATCH, GATE3, HID, GEMM_ABT, 0, 0);
        gru_gate<<<cdiv(NH, B256), B256, 0, st>>>(GI, GH, h, h_bf,
                                                  hseq0_bf + (size_t)t * NH);
    }
    // ================= GRU layer 1 =================
    zero_f32<<<cdiv(NH, B256), B256, 0, st>>>(h, NH);
    zero_bf <<<cdiv(NH, B256), B256, 0, st>>>(h_bf, NH);
    for (int t = 0; t < TSTEPS; ++t) {
        gemm(st, hseq0_bf + (size_t)t * NH, W1ih_bf, GI, nullptr, b1ih,
             NBATCH, GATE3, HID, GEMM_ABT, 0, 0);
        gemm(st, h_bf, W1hh_bf, GH, nullptr, b1hh, NBATCH, GATE3, HID, GEMM_ABT, 0, 0);
        gru_gate<<<cdiv(NH, B256), B256, 0, st>>>(GI, GH, h, h_bf, nullptr);
    }
    float* x_h   = h;      // (N,H) final hidden
    bf16*  x_h_bf = h_bf;

    // ================= concept attention =================
    col_sum<<<CCON, B256, 0, st>>>(concept, csum, NBATCH, CCON);
    s2c_kernel<<<cdiv(NBATCH * CCON, B256), B256, 0, st>>>(concept, csum, s2c_bf);
    gemm(st, s2c_bf, x_h_bf, hidden0, hidden0_bf, nullptr, CCON, HID, NBATCH, GEMM_ATB, 0, 0);
    row_nonzero<<<cdiv(CCON, B256), B256, 0, st>>>(hidden0, valid0, CCON, HID);
    gemm(st, x_h_bf, hidden0_bf, NC_a, nullptr, nullptr, NBATCH, CCON, HID, GEMM_ABT, 0, 0);
    col_softmax<<<CCON, B256, 0, st>>>(NC_a, NC_b, NBATCH, CCON);
    f2bf<<<cdiv(NBATCH * CCON, B256), B256, 0, st>>>(NC_b, NC_bf, NBATCH * CCON);
    gemm(st, NC_bf, x_h_bf, hidden1, hidden1_bf, nullptr, CCON, HID, NBATCH, GEMM_ATB, 0, 0);
    row_l2norm<<<cdiv(NBATCH, B256), B256, 0, st>>>(x_h, norm1, NBATCH, HID);
    row_l2norm<<<cdiv(CCON, B256), B256, 0, st>>>(hidden1, norm2, CCON, HID);
    gemm(st, x_h_bf, hidden1_bf, NC_a, nullptr, nullptr, NBATCH, CCON, HID, GEMM_ABT, 0, 0);
    cos_finish<<<cdiv(NBATCH * CCON, B256), B256, 0, st>>>(NC_a, norm1, norm2, NBATCH, CCON);
    row_softmax<<<NBATCH, B256, 0, st>>>(NC_a, valid0, NC_bf, NBATCH, CCON);
    gemm(st, NC_bf, hidden1_bf, tmpH, tmpH_bf, nullptr, NBATCH, HID, CCON, GEMM_AB, 0, 0);
    gemm(st, tmpH_bf, fcw_bf[0], p_shared, p_shared_bf, fc_ps_b, NBATCH, HID, HID, GEMM_ABT, 0, 0);

    // read_mem(p_shared, m0)
    gemm(st, p_shared_bf, m0_bf, score, nullptr, nullptr, NBATCH, MMEM, HID, GEMM_ABT, 0, 0);
    row_softmax<<<NBATCH, B256, 0, st>>>(score, nullptr, score_bf, NBATCH, MMEM);
    gemm(st, score_bf, m0_bf, Rtmp, nullptr, nullptr, NBATCH, HID, MMEM, GEMM_AB, 0, 0);
    ew_mul_inplace<<<cdiv(NH, B256), B256, 0, st>>>(p_shared, Rtmp, p_shared_bf, NH);

    // upload_mem(p_shared, m0) -> out_mem0
    gemm(st, p_shared_bf, m0_bf, score, nullptr, nullptr, NBATCH, MMEM, HID, GEMM_ABT, 0, 0);
    row_argmax<<<cdiv(NBATCH, B256), B256, 0, st>>>(score, gidx, NBATCH, MMEM);
    col_softmax<<<MMEM, B256, 0, st>>>(score, sq, NBATCH, MMEM);
    col_max<<<MMEM, B256, 0, st>>>(sq, mx, NBATCH, MMEM);
    upload_w_kernel<<<cdiv(NBATCH, B256), B256, 0, st>>>(sq, gidx, mx, wv);
    zero_f32<<<cdiv(MMEM * HID, B256), B256, 0, st>>>(qu, MMEM * HID);
    scatter_qu<<<cdiv(NH, B256), B256, 0, st>>>(p_shared, wv, gidx, qu);
    mem_finalize<<<1, MMEM, 0, st>>>(qu, m0, out_mem0);

    // p_back / output_ps
    gemm(st, p_shared_bf, fcw_bf[4], p_back, nullptr, fc_ps_back_b, NBATCH, HID, HID, GEMM_ABT, 0, 0);
    zero_f32<<<cdiv(NH, B256), B256, 0, st>>>(all_info, NH);
    gemm(st, p_shared_bf, fcw_bf[2], all_info, nullptr, fc_ps_fore_b, NBATCH, HID, HID, GEMM_ABT, 1, 1);

    // ================= h_si graph attention =================
    ew_sub<<<cdiv(NH, B256), B256, 0, st>>>(x_h, p_back, h_si, h_si_bf, NH);
    row_l2norm<<<cdiv(NBATCH, B256), B256, 0, st>>>(h_si, norm1, NBATCH, HID);
    gemm(st, h_si_bf, h_si_bf, NN, nullptr, nullptr, NBATCH, NBATCH, HID, GEMM_ABT, 0, 0);
    cos_finish<<<cdiv(NBATCH * NBATCH, B256), B256, 0, st>>>(NN, norm1, norm1, NBATCH, NBATCH);
    zero_f32<<<cdiv(NBATCH, B256), B256, 0, st>>>(colsum, NBATCH);
    topk3_kernel<<<cdiv(NBATCH, B256), B256, 0, st>>>(NN, diag, tv, tc, colsum);
    zero_f32<<<cdiv(NH, B256), B256, 0, st>>>(hid2, NH);
    scatter_hid2<<<cdiv(NBATCH * TOPK3 * HID, B256), B256, 0, st>>>(tv, tc, h_si, hid2);
    hid2_diag<<<cdiv(NH, B256), B256, 0, st>>>(colsum, diag, h_si, hid2);
    f2bf<<<cdiv(NH, B256), B256, 0, st>>>(hid2, hid2_bf, NH);
    row_nonzero<<<cdiv(NBATCH, B256), B256, 0, st>>>(hid2, validh, NBATCH, HID);
    row_l2norm<<<cdiv(NBATCH, B256), B256, 0, st>>>(hid2, norm2, NBATCH, HID);
    gemm(st, h_si_bf, hid2_bf, NN, nullptr, nullptr, NBATCH, NBATCH, HID, GEMM_ABT, 0, 0);
    cos_finish<<<cdiv(NBATCH * NBATCH, B256), B256, 0, st>>>(NN, norm1, norm2, NBATCH, NBATCH);
    row_softmax<<<NBATCH, B256, 0, st>>>(NN, validh, NN_bf, NBATCH, NBATCH);
    gemm(st, NN_bf, hid2_bf, tmpH, tmpH_bf, nullptr, NBATCH, HID, NBATCH, GEMM_AB, 0, 0);
    gemm(st, tmpH_bf, fcw_bf[1], h_shared, h_shared_bf, fc_hs_b, NBATCH, HID, HID, GEMM_ABT, 0, 0);

    // read_mem(h_shared, m1)
    gemm(st, h_shared_bf, m1_bf, score, nullptr, nullptr, NBATCH, MMEM, HID, GEMM_ABT, 0, 0);
    row_softmax<<<NBATCH, B256, 0, st>>>(score, nullptr, score_bf, NBATCH, MMEM);
    gemm(st, score_bf, m1_bf, Rtmp, nullptr, nullptr, NBATCH, HID, MMEM, GEMM_AB, 0, 0);
    ew_mul_inplace<<<cdiv(NH, B256), B256, 0, st>>>(h_shared, Rtmp, h_shared_bf, NH);

    // upload_mem(h_shared, m1) -> out_mem1
    gemm(st, h_shared_bf, m1_bf, score, nullptr, nullptr, NBATCH, MMEM, HID, GEMM_ABT, 0, 0);
    row_argmax<<<cdiv(NBATCH, B256), B256, 0, st>>>(score, gidx, NBATCH, MMEM);
    col_softmax<<<MMEM, B256, 0, st>>>(score, sq, NBATCH, MMEM);
    col_max<<<MMEM, B256, 0, st>>>(sq, mx, NBATCH, MMEM);
    upload_w_kernel<<<cdiv(NBATCH, B256), B256, 0, st>>>(sq, gidx, mx, wv);
    zero_f32<<<cdiv(MMEM * HID, B256), B256, 0, st>>>(qu, MMEM * HID);
    scatter_qu<<<cdiv(NH, B256), B256, 0, st>>>(h_shared, wv, gidx, qu);
    mem_finalize<<<1, MMEM, 0, st>>>(qu, m1, out_mem1);

    // h_back / output_hs / output_indi
    gemm(st, h_shared_bf, fcw_bf[5], h_back, nullptr, fc_hs_back_b, NBATCH, HID, HID, GEMM_ABT, 0, 0);
    gemm(st, h_shared_bf, fcw_bf[3], all_info, nullptr, fc_hs_fore_b, NBATCH, HID, HID, GEMM_ABT, 1, 1);
    ew_sub3<<<cdiv(NH, B256), B256, 0, st>>>(x_h, p_back, h_back, tmpH, tmpH_bf, NH);
    gemm(st, tmpH_bf, fcw_bf[6], all_info, nullptr, fc_indi_b, NBATCH, HID, HID, GEMM_ABT, 1, 1);

    // final prediction
    pred_kernel<<<cdiv(NBATCH, B256), B256, 0, st>>>(all_info, fc_out_w, fc_out_b, out_pred);
}
